// LlamaAttention_47528108097834
// MI455X (gfx1250) — matrix operand order repaired and checked by reading, not compiled, not run on hardware
//
#include <hip/hip_runtime.h>
#include <math.h>

// ---------------- constants ----------------
#define SEQ    4096
#define HID    1024
#define NH     16
#define HD     64          // head dim
#define NEGBIG (-1.0e30f)

typedef __attribute__((ext_vector_type(16))) __bf16 v16bf;
typedef __attribute__((ext_vector_type(8)))  __bf16 v8bf;
typedef __attribute__((ext_vector_type(8)))  float  v8f;

// ---------------- WMMA helpers ----------------
static __device__ __forceinline__ v16bf combine16(v8bf lo, v8bf hi) {
    return __builtin_shufflevector(lo, hi, 0,1,2,3,4,5,6,7,8,9,10,11,12,13,14,15);
}

// A fragment (16x32 bf16, MxK): lane holds row = lane%16,
// K values { 8g..8g+7 , 8g+16..8g+23 } with g = lane/16.  (ISA 7.12.2)
static __device__ __forceinline__ v16bf frag_a(const __bf16* base, int ld, int lane) {
    const __bf16* p = base + (lane & 15) * ld + ((lane >> 4) << 3);
    v8bf lo = *(const v8bf*)(p);
    v8bf hi = *(const v8bf*)(p + 16);
    return combine16(lo, hi);
}

// B fragment (32x16 bf16, KxN), stored as rows of length-K (B^T / weight rows):
// lane holds col = lane%16, K values { 8g..8g+7 , 8g+16..8g+23 } with g = lane/16.
static __device__ __forceinline__ v16bf frag_b(const __bf16* base, int ld, int lane) {
    const __bf16* p = base + (lane & 15) * ld + ((lane >> 4) << 3);
    v8bf lo = *(const v8bf*)(p);
    v8bf hi = *(const v8bf*)(p + 16);
    return combine16(lo, hi);
}

static __device__ __forceinline__ v8f wmma_bf16(v16bf a, v16bf b, v8f c) {
    return __builtin_amdgcn_wmma_f32_16x16x32_bf16(false, a, false, b, (short)0, c, false, false);
}

// ---------------- CDNA5 async global->LDS copy (ASYNCcnt path) ----------------
static __device__ __forceinline__ void async_cp16(void* lds_dst, const void* gsrc) {
    unsigned           l = (unsigned)(unsigned long long)lds_dst;
    unsigned long long g = (unsigned long long)gsrc;
    asm volatile("global_load_async_to_lds_b128 %0, %1, off"
                 :: "v"(l), "v"(g) : "memory");
}
static __device__ __forceinline__ void async_wait0() {
    asm volatile("s_wait_asynccnt 0x0" ::: "memory");
}

// =====================================================================
// Kernel 1: Q/K/V = X @ W^T  (bf16 WMMA, f32 accumulate), fused RoPE.
// Q,K stored per-head [h][s][64]; V stored TRANSPOSED per head [h][d][s]
// so the attention kernel can stage V^T tiles with contiguous async copies.
// grid = (SEQ/64, HID/64, 3), block 256.
// =====================================================================
__global__ __launch_bounds__(256)
void qkv_rope_kernel(const float* __restrict__ X,
                     const float* __restrict__ Wq,
                     const float* __restrict__ Wk,
                     const float* __restrict__ Wv,
                     __bf16* __restrict__ Qb,
                     __bf16* __restrict__ Kb,
                     __bf16* __restrict__ Vb) {
    __shared__ __align__(16) __bf16 Xs[64 * 32];
    __shared__ __align__(16) __bf16 Ws[64 * 32];
    __shared__ __align__(16) float  Cs[64 * 64];

    const int tid  = threadIdx.x;
    const int lane = tid & 31;
    const int wv   = tid >> 5;          // 0..7
    const int s0   = blockIdx.x * 64;
    const int n0   = blockIdx.y * 64;   // 64 cols == one head
    const int mat  = blockIdx.z;        // 0=Q 1=K 2=V
    const float* W = (mat == 0) ? Wq : (mat == 1) ? Wk : Wv;

    const int mi0 = wv >> 2;            // m-subtiles mi0, mi0+2
    const int nj  = wv & 3;             // n-subtile
    v8f acc0 = {}, acc1 = {};

    const int ldr = tid >> 2;           // staging row 0..63
    const int ldc = (tid & 3) * 8;      // staging col0

    for (int k0 = 0; k0 < HID; k0 += 32) {
        const float* xp = X + (size_t)(s0 + ldr) * HID + k0 + ldc;
        const float* wp = W + (size_t)(n0 + ldr) * HID + k0 + ldc;
        if (k0 + 32 < HID) {            // next k-slice -> global_prefetch_b8 (WGP scope)
            __builtin_prefetch(xp + 32, 0, 3);
            __builtin_prefetch(wp + 32, 0, 3);
        }
#pragma unroll
        for (int i = 0; i < 8; ++i) {
            Xs[ldr * 32 + ldc + i] = (__bf16)xp[i];
            Ws[ldr * 32 + ldc + i] = (__bf16)wp[i];
        }
        __syncthreads();

        v16bf b  = frag_b(&Ws[nj * 16 * 32], 32, lane);
        v16bf a0 = frag_a(&Xs[(mi0 * 16) * 32], 32, lane);
        v16bf a1 = frag_a(&Xs[((mi0 + 2) * 16) * 32], 32, lane);
        acc0 = wmma_bf16(a0, b, acc0);
        acc1 = wmma_bf16(a1, b, acc1);
        __syncthreads();
    }

    // spill C (f32) so RoPE pairs (d, d+32) are addressable by any thread
    {
        const int col = nj * 16 + (lane & 15);
        const int rb0 = mi0 * 16 + 8 * (lane >> 4);
        const int rb1 = (mi0 + 2) * 16 + 8 * (lane >> 4);
#pragma unroll
        for (int v = 0; v < 8; ++v) {
            Cs[(rb0 + v) * 64 + col] = acc0[v];
            Cs[(rb1 + v) * 64 + col] = acc1[v];
        }
    }
    __syncthreads();

    const int h  = blockIdx.y;
    const int r  = tid >> 2;            // 0..63
    const int c0 = (tid & 3) * 16;      // 16 cols each
    const int s  = s0 + r;
    if (mat < 2) {
        __bf16* dst = (mat == 0) ? Qb : Kb;
#pragma unroll
        for (int i = 0; i < 16; ++i) {
            const int d = c0 + i;
            const int j = d & 31;                       // rope pair index
            const float theta = __expf((float)j * -0.2878231366242557f); // -ln(1e4)/32
            float sn, cs;
            __sincosf((float)s * theta, &sn, &cs);
            const float e1 = Cs[r * 64 + j];
            const float e2 = Cs[r * 64 + j + 32];
            const float val = (d < 32) ? (e1 * cs - e2 * sn) : (e1 * sn + e2 * cs);
            dst[((size_t)h * SEQ + s) * HD + d] = (__bf16)val;
        }
    } else {
        // V transposed: Vb[h][d][s]
#pragma unroll
        for (int i = 0; i < 16; ++i) {
            const int d = c0 + i;
            Vb[((size_t)h * HD + d) * SEQ + s] = (__bf16)Cs[r * 64 + d];
        }
    }
}

// =====================================================================
// Kernel 2: causal flash attention per head.
// block = 128 q-rows x 1 head; wave owns a 16-row strip (in-wave softmax).
// K / V^T tiles DOUBLE-BUFFERED: async copies for tile kt+1 are issued
// before computing tile kt, so the ASYNCcnt copy hides under WMMA+softmax.
// One barrier per tile. grid = (SEQ/128, NH), block 256. LDS = 64KB.
// =====================================================================
__global__ __launch_bounds__(256)
void attn_kernel(const __bf16* __restrict__ Qb,
                 const __bf16* __restrict__ Kb,
                 const __bf16* __restrict__ Vb,
                 __bf16* __restrict__ ctx) {
    __shared__ __align__(16) __bf16 Qs[128 * 64];
    __shared__ __align__(16) __bf16 Ks[2][64 * 64];
    __shared__ __align__(16) __bf16 Vts[2][64 * 64];   // V^T tile: [d][n]
    __shared__ __align__(16) __bf16 Ps[8 * 16 * 64];   // per-wave P strip

    const int tid  = threadIdx.x;
    const int lane = tid & 31;
    const int wv   = tid >> 5;
    const int h    = blockIdx.y;
    const int q0   = blockIdx.x * 128;

    const __bf16* Qh = Qb + (size_t)h * SEQ * HD;     // [s][d]
    const __bf16* Kh = Kb + (size_t)h * SEQ * HD;     // [s][d]
    const __bf16* Vh = Vb + (size_t)h * HD * SEQ;     // [d][s]  (pre-transposed)

    const int sr = tid >> 2;            // staging row 0..63
    const int sc = (tid & 3) * 16;      // staging col0 (two 16B chunks per row)

    // stage Q strip (128x64): async 16B copies, 4 per thread
    for (int c = tid; c < 1024; c += 256) {
        const int row = c >> 3, col = (c & 7) * 8;
        async_cp16(&Qs[row * 64 + col], &Qh[(size_t)(q0 + row) * HD + col]);
    }
    // stage K/V^T tile 0 into buffer 0
    {
        async_cp16(&Ks[0][sr * 64 + sc],      &Kh[(size_t)sr * HD + sc]);
        async_cp16(&Ks[0][sr * 64 + sc + 8],  &Kh[(size_t)sr * HD + sc + 8]);
        async_cp16(&Vts[0][sr * 64 + sc],     &Vh[(size_t)sr * SEQ + sc]);
        async_cp16(&Vts[0][sr * 64 + sc + 8], &Vh[(size_t)sr * SEQ + sc + 8]);
    }
    async_wait0();
    __syncthreads();

    float m_run[8], l_run[8];
    v8f acco[4] = {};
#pragma unroll
    for (int v = 0; v < 8; ++v) { m_run[v] = NEGBIG; l_run[v] = 0.f; }

    const int row_in = 8 * (lane >> 4);         // +v -> row within strip
    const int qrow   = q0 + wv * 16 + row_in;   // +v -> global q row
    const int nkt    = blockIdx.x * 2 + 2;      // causal: tiles through the diagonal

    for (int kt = 0; kt < nkt; ++kt) {
        const int n0g = kt * 64;
        const int cur = kt & 1;

        // issue async copies for NEXT tile into the other buffer (overlap w/ compute)
        if (kt + 1 < nkt) {
            const int nxt  = (kt + 1) & 1;
            const int n1g  = n0g + 64;
            async_cp16(&Ks[nxt][sr * 64 + sc],      &Kh[(size_t)(n1g + sr) * HD + sc]);
            async_cp16(&Ks[nxt][sr * 64 + sc + 8],  &Kh[(size_t)(n1g + sr) * HD + sc + 8]);
            async_cp16(&Vts[nxt][sr * 64 + sc],     &Vh[(size_t)sr * SEQ + n1g + sc]);
            async_cp16(&Vts[nxt][sr * 64 + sc + 8], &Vh[(size_t)sr * SEQ + n1g + sc + 8]);
        }

        // S = Q K^T (16x64 per wave)
        v8f accs[4] = {};
#pragma unroll
        for (int dk = 0; dk < 2; ++dk) {
            v16bf a = frag_a(&Qs[wv * 16 * 64 + dk * 32], 64, lane);
#pragma unroll
            for (int j = 0; j < 4; ++j) {
                v16bf b = frag_b(&Ks[cur][j * 16 * 64 + dk * 32], 64, lane);
                accs[j] = wmma_bf16(a, b, accs[j]);
            }
        }

        // scale + causal mask + online softmax (rows live in half-waves)
#pragma unroll
        for (int v = 0; v < 8; ++v) {
            const int q = qrow + v;
            float mx = NEGBIG;
#pragma unroll
            for (int j = 0; j < 4; ++j) {
                const int kcol = n0g + j * 16 + (lane & 15);
                float sv = accs[j][v] * 0.125f;             // 1/sqrt(64)
                if (kcol > q) sv = NEGBIG;
                accs[j][v] = sv;
                mx = fmaxf(mx, sv);
            }
#pragma unroll
            for (int m = 1; m < 16; m <<= 1) mx = fmaxf(mx, __shfl_xor(mx, m, 32));
            const float mn    = fmaxf(m_run[v], mx);
            const float alpha = __expf(m_run[v] - mn);
            float rs = 0.f;
#pragma unroll
            for (int j = 0; j < 4; ++j) {
                const float pv = __expf(accs[j][v] - mn);
                accs[j][v] = pv;                            // reuse accs as P
                rs += pv;
            }
#pragma unroll
            for (int m = 1; m < 16; m <<= 1) rs += __shfl_xor(rs, m, 32);
            l_run[v] = l_run[v] * alpha + rs;
            m_run[v] = mn;
#pragma unroll
            for (int jd = 0; jd < 4; ++jd) acco[jd][v] = acco[jd][v] * alpha;
        }

        // P -> per-wave LDS (bf16); same wave reads it back as A fragments
#pragma unroll
        for (int j = 0; j < 4; ++j)
#pragma unroll
            for (int v = 0; v < 8; ++v)
                Ps[wv * 1024 + (row_in + v) * 64 + j * 16 + (lane & 15)] = (__bf16)accs[j][v];

        // O += P V
#pragma unroll
        for (int nk = 0; nk < 2; ++nk) {
            v16bf a = frag_a(&Ps[wv * 1024 + nk * 32], 64, lane);
#pragma unroll
            for (int jd = 0; jd < 4; ++jd) {
                v16bf b = frag_b(&Vts[cur][jd * 16 * 64 + nk * 32], 64, lane);
                acco[jd] = wmma_bf16(a, b, acco[jd]);
            }
        }

        async_wait0();      // next tile's copies landed (overlapped with the above)
        __syncthreads();    // publish buffers / all waves done with current tile
    }

    // normalize + store context [s][h*64+d] as bf16 for the O-projection
#pragma unroll
    for (int v = 0; v < 8; ++v) {
        const float inv  = 1.0f / l_run[v];
        const int   srow = qrow + v;
#pragma unroll
        for (int jd = 0; jd < 4; ++jd) {
            const int col = h * HD + jd * 16 + (lane & 15);
            ctx[(size_t)srow * HID + col] = (__bf16)(acco[jd][v] * inv);
        }
    }
}

// =====================================================================
// Kernel 3: out = ctx @ Wo^T + bo   (f32 out). grid = (SEQ/64, HID/64).
// ctx tile staged with async copies; Wo converted f32->bf16 through VGPRs.
// =====================================================================
__global__ __launch_bounds__(256)
void oproj_kernel(const __bf16* __restrict__ ctx,
                  const float* __restrict__ Wo,
                  const float* __restrict__ bo,
                  float* __restrict__ out) {
    __shared__ __align__(16) __bf16 As[64 * 32];
    __shared__ __align__(16) __bf16 Ws[64 * 32];

    const int tid  = threadIdx.x;
    const int lane = tid & 31;
    const int wv   = tid >> 5;
    const int s0   = blockIdx.x * 64;
    const int n0   = blockIdx.y * 64;
    const int mi0  = wv >> 2;
    const int nj   = wv & 3;
    v8f acc0 = {}, acc1 = {};

    const int ldr = tid >> 2;
    const int ldc = (tid & 3) * 8;

    for (int k0 = 0; k0 < HID; k0 += 32) {
        async_cp16(&As[ldr * 32 + ldc], &ctx[(size_t)(s0 + ldr) * HID + k0 + ldc]);
        const float* wp = Wo + (size_t)(n0 + ldr) * HID + k0 + ldc;
        if (k0 + 32 < HID) __builtin_prefetch(wp + 32, 0, 3);
#pragma unroll
        for (int i = 0; i < 8; ++i) Ws[ldr * 32 + ldc + i] = (__bf16)wp[i];
        async_wait0();
        __syncthreads();

        v16bf b  = frag_b(&Ws[nj * 16 * 32], 32, lane);
        v16bf a0 = frag_a(&As[(mi0 * 16) * 32], 32, lane);
        v16bf a1 = frag_a(&As[((mi0 + 2) * 16) * 32], 32, lane);
        acc0 = wmma_bf16(a0, b, acc0);
        acc1 = wmma_bf16(a1, b, acc1);
        __syncthreads();
    }

    const int col    = n0 + nj * 16 + (lane & 15);
    const float bias = bo[col];
    const int rb0 = s0 + mi0 * 16 + 8 * (lane >> 4);
    const int rb1 = s0 + (mi0 + 2) * 16 + 8 * (lane >> 4);
#pragma unroll
    for (int v = 0; v < 8; ++v) {
        out[(size_t)(rb0 + v) * HID + col] = acc0[v] + bias;
        out[(size_t)(rb1 + v) * HID + col] = acc1[v] + bias;
    }
}

// =====================================================================
// Launch: inputs = { X, position_ids, mask, Wq, Wk, Wv, Wo, bo }
// =====================================================================
extern "C" void kernel_launch(void* const* d_in, const int* in_sizes, int n_in,
                              void* d_out, int out_size, void* d_ws, size_t ws_size,
                              hipStream_t stream) {
    (void)in_sizes; (void)n_in; (void)out_size; (void)ws_size;

    const float* X  = (const float*)d_in[0];
    // d_in[1] = position_ids (arange, implied), d_in[2] = causal mask (implied)
    const float* Wq = (const float*)d_in[3];
    const float* Wk = (const float*)d_in[4];
    const float* Wv = (const float*)d_in[5];
    const float* Wo = (const float*)d_in[6];
    const float* bo = (const float*)d_in[7];
    float* out = (float*)d_out;

    // workspace: Q,K (per-head [h][s][64]) + V^T (per-head [h][d][s]) + ctx
    const size_t per = (size_t)NH * SEQ * HD;     // 4M elements each
    __bf16* Qb  = (__bf16*)d_ws;
    __bf16* Kb  = Qb + per;
    __bf16* Vb  = Kb + per;
    __bf16* ctx = Vb + per;

    dim3 g1(SEQ / 64, HID / 64, 3);
    qkv_rope_kernel<<<g1, 256, 0, stream>>>(X, Wq, Wk, Wv, Qb, Kb, Vb);

    dim3 g2(SEQ / 128, NH);
    attn_kernel<<<g2, 256, 0, stream>>>(Qb, Kb, Vb, ctx);

    dim3 g3(SEQ / 64, HID / 64);
    oproj_kernel<<<g3, 256, 0, stream>>>(ctx, Wo, bo, out);
}
